// Model_72541997629504
// MI455X (gfx1250) — compile-verified, matrix operand order checked by reading, not run on hardware
//
#include <hip/hip_runtime.h>
#include <hip/hip_bf16.h>

// ---------------- problem constants (from reference setup_inputs) -----------
#define NN    20000     // nodes
#define EE    320000    // edges
#define FIN   8710      // input features
#define FH    1024      // hidden features
#define FOUT  70        // output classes

// ---------------- WMMA GEMM tiling ------------------------------------------
#define BM 128
#define BN 64
#define BK 32
#define LDA_S 40        // LDS row stride for A (bf16 elems), padded
#define LDB_S 40        // LDS row stride for B^T (bf16 elems), padded

typedef __attribute__((ext_vector_type(16))) __bf16 v16bf;
typedef __attribute__((ext_vector_type(8)))  float  v8f;

union FragBF16 {
    v16bf v;
    unsigned int u[8];
};

__device__ __forceinline__ unsigned int fbits(float f) {
    union { float f; unsigned int u; } c; c.f = f; return c.u;
}

// f32 -> bf16 (round to nearest), scalar
__device__ __forceinline__ unsigned short f2bf(float f) {
    return (unsigned short)((fbits(f) + 0x8000u) >> 16);
}

// pack two f32 -> two bf16 in one dword: 2x v_add + 1x v_perm_b32
__device__ __forceinline__ unsigned int pack2bf(float a, float b) {
    const unsigned int ua = fbits(a) + 0x8000u;
    const unsigned int ub = fbits(b) + 0x8000u;
    // result bytes [1:0] = ua[3:2], [3:2] = ub[3:2]
    return __builtin_amdgcn_perm(ub, ua, 0x07060302u);
}

// C[M x Ncols] = A[M x K] (f32) * B[K x Ncols] (f32), bf16 WMMA, f32 accumulate
// grid: (N-tiles, M-tiles)  -- N fastest so concurrent blocks share A in L2
__global__ __launch_bounds__(256) void gemm_wmma_bf16(
    const float* __restrict__ A, const float* __restrict__ B,
    float* __restrict__ C, int M, int K, int Ncols)
{
    // double-buffered staging tiles
    __shared__ unsigned short As[2][BM * LDA_S];   // [m][k] bf16
    __shared__ unsigned short Bs[2][BN * LDB_S];   // [n][k] bf16 (transposed)

    const int tid  = threadIdx.x;
    const int lane = tid & 31;
    const int wid  = tid >> 5;             // 0..7
    const int waveM = (wid & 3) * 32;      // 0,32,64,96
    const int waveN = (wid >> 2) * 32;     // 0,32

    const int blockM = blockIdx.y * BM;    // M on y
    const int blockN = blockIdx.x * BN;    // N on x (fastest-varying)

    const int half = lane >> 4;            // 0: K 0-7/16-23 ; 1: K 8-15/24-31
    const int l16  = lane & 15;

    // A staging map: 8 threads per 32-wide row; each thread 4 rows, 4 cols
    const int ar = tid >> 3;               // 0..31
    const int ac = (tid & 7) * 4;          // 0..28
    // B staging map: n = tid&63, k-pair index = (tid>>6) + 4r
    const int bn  = tid & 63;
    const int bkp = tid >> 6;              // 0..3

    v8f acc[2][2] = {};

    auto stage = [&](int buf, int k0) {
        unsigned short* as = &As[buf][0];
        unsigned short* bs = &Bs[buf][0];

        // ----------------- A tile -----------------
        if (blockM + BM <= M && k0 + BK <= K) {
            // fast path: unguarded vector loads (row base 8B-aligned: K even)
            #pragma unroll
            for (int r = 0; r < 4; ++r) {
                const int  lm   = ar + 32 * r;
                const long base = (long)(blockM + lm) * K + k0 + ac;
                const float2 v0 = *(const float2*)(A + base);
                const float2 v1 = *(const float2*)(A + base + 2);
                if (k0 + BK < K) __builtin_prefetch(A + base + BK, 0, 3);
                uint2 p;
                p.x = pack2bf(v0.x, v0.y);
                p.y = pack2bf(v1.x, v1.y);
                *(uint2*)&as[lm * LDA_S + ac] = p;   // ds_store_b64
            }
        } else {
            // slow path: clamped loads + select (branch-free)
            #pragma unroll
            for (int r = 0; r < 4; ++r) {
                const int lm  = ar + 32 * r;
                const int gm  = blockM + lm;
                const int gmc = gm < M - 1 ? gm : M - 1;
                #pragma unroll
                for (int j = 0; j < 4; ++j) {
                    const int gk  = k0 + ac + j;
                    const int gkc = gk < K - 1 ? gk : K - 1;
                    float v = A[(long)gmc * K + gkc];
                    v = (gm < M && gk < K) ? v : 0.0f;
                    as[lm * LDA_S + ac + j] = f2bf(v);
                }
            }
        }

        // ----------------- B tile (transposed stage) -----------------
        if (blockN + BN <= Ncols && k0 + BK <= K) {
            #pragma unroll
            for (int r = 0; r < 4; ++r) {
                const int  kp   = bkp + 4 * r;       // 0..15
                const int  kk   = kp * 2;
                const long base = (long)(k0 + kk) * Ncols + blockN + bn;
                const float v0  = B[base];           // coalesced across lanes
                const float v1  = B[base + Ncols];
                *(unsigned int*)&bs[bn * LDB_S + kk] = pack2bf(v0, v1);
            }
        } else {
            #pragma unroll
            for (int r = 0; r < 4; ++r) {
                const int kp  = bkp + 4 * r;
                const int kk  = kp * 2;
                const int gn  = blockN + bn;
                const int gnc = gn < Ncols - 1 ? gn : Ncols - 1;
                float va[2];
                #pragma unroll
                for (int j = 0; j < 2; ++j) {
                    const int gk  = k0 + kk + j;
                    const int gkc = gk < K - 1 ? gk : K - 1;
                    float v = B[(long)gkc * Ncols + gnc];
                    va[j] = (gk < K && gn < Ncols) ? v : 0.0f;
                }
                *(unsigned int*)&bs[bn * LDB_S + kk] = pack2bf(va[0], va[1]);
            }
        }
    };

    const int nk = (K + BK - 1) / BK;
    stage(0, 0);

    for (int t = 0; t < nk; ++t) {
        __syncthreads();                       // buffer (t&1) is ready
        if (t + 1 < nk) stage((t + 1) & 1, (t + 1) * BK);

        const unsigned short* as = &As[t & 1][0];
        const unsigned short* bs = &Bs[t & 1][0];

        FragBF16 af[2], bfr[2];
        #pragma unroll
        for (int p = 0; p < 8; ++p) {
            const int kk = ((p < 4) ? (2 * p) : (2 * p + 8)) + half * 8;
            af[0].u[p]  = *(const unsigned int*)&as[(waveM +      l16) * LDA_S + kk];
            af[1].u[p]  = *(const unsigned int*)&as[(waveM + 16 + l16) * LDA_S + kk];
            bfr[0].u[p] = *(const unsigned int*)&bs[(waveN +      l16) * LDB_S + kk];
            bfr[1].u[p] = *(const unsigned int*)&bs[(waveN + 16 + l16) * LDB_S + kk];
        }
        #pragma unroll
        for (int r = 0; r < 2; ++r)
            #pragma unroll
            for (int c = 0; c < 2; ++c)
                acc[r][c] = __builtin_amdgcn_wmma_f32_16x16x32_bf16(
                    false, af[r].v, false, bfr[c].v,
                    (short)0, acc[r][c], false, false);
    }

    // store C: 16x16 f32 D layout: VGPR v -> M = v + 8*half, N = l16
    #pragma unroll
    for (int r = 0; r < 2; ++r) {
        #pragma unroll
        for (int c = 0; c < 2; ++c) {
            const int col = blockN + waveN + c * 16 + l16;
            if (col >= Ncols) continue;
            const int row0 = blockM + waveM + r * 16 + half * 8;
            #pragma unroll
            for (int v = 0; v < 8; ++v) {
                const int row = row0 + v;
                if (row < M) C[(long)row * Ncols + col] = acc[r][c][v];
            }
        }
    }
}

// ---------------- graph-norm kernels -----------------------------------------
__global__ void k_deg_init(float* deg, int n) {
    int i = blockIdx.x * blockDim.x + threadIdx.x;
    if (i < n) deg[i] = 1.0f;   // self-loop contribution
}

__global__ void k_deg_acc(const long long* __restrict__ edges, float* deg, int E) {
    int e = blockIdx.x * blockDim.x + threadIdx.x;
    if (e < E) {
        int dst = (int)edges[(long)E + e];
        __hip_atomic_fetch_add(&deg[dst], 1.0f, __ATOMIC_RELAXED, __HIP_MEMORY_SCOPE_AGENT);
    }
}

__global__ void k_dinv(float* deg, int n) {
    int i = blockIdx.x * blockDim.x + threadIdx.x;
    if (i < n) {
        float d = deg[i];
        deg[i] = (d > 0.0f) ? rsqrtf(d) : 0.0f;
    }
}

__global__ void k_norm(const long long* __restrict__ edges,
                       const float* __restrict__ dinv,
                       float* __restrict__ norm, int E, int n) {
    int e = blockIdx.x * blockDim.x + threadIdx.x;
    if (e >= E + n) return;
    int s, d;
    if (e < E) { s = (int)edges[e]; d = (int)edges[(long)E + e]; }
    else       { s = d = e - E; }
    norm[e] = dinv[s] * dinv[d];
}

// ---------------- scatter-add message passing --------------------------------
// F == 4*blockDim variant (F=1024, block=256): vector loads, hw float atomics
__global__ __launch_bounds__(256) void k_scatter_vec4(
    const float* __restrict__ h, const long long* __restrict__ edges,
    const float* __restrict__ norm, float* __restrict__ agg, int E, int n)
{
    const int e = blockIdx.x;
    int src, dst;
    if (e < E) { src = (int)edges[e]; dst = (int)edges[(long)E + e]; }
    else       { src = dst = e - E; }
    const float w = norm[e];
    const int f0 = threadIdx.x * 4;
    const float4 v = *(const float4*)(h + (long)src * FH + f0);
    float* ad = agg + (long)dst * FH + f0;
    __hip_atomic_fetch_add(&ad[0], v.x * w, __ATOMIC_RELAXED, __HIP_MEMORY_SCOPE_AGENT);
    __hip_atomic_fetch_add(&ad[1], v.y * w, __ATOMIC_RELAXED, __HIP_MEMORY_SCOPE_AGENT);
    __hip_atomic_fetch_add(&ad[2], v.z * w, __ATOMIC_RELAXED, __HIP_MEMORY_SCOPE_AGENT);
    __hip_atomic_fetch_add(&ad[3], v.w * w, __ATOMIC_RELAXED, __HIP_MEMORY_SCOPE_AGENT);
}

__global__ void k_scatter(const float* __restrict__ h,
                          const long long* __restrict__ edges,
                          const float* __restrict__ norm,
                          float* __restrict__ agg,
                          int E, int n, int F)
{
    const int e = blockIdx.x;
    int src, dst;
    if (e < E) { src = (int)edges[e]; dst = (int)edges[(long)E + e]; }
    else       { src = dst = e - E; }
    const float w = norm[e];
    const float* hs = h + (long)src * F;
    float* ad = agg + (long)dst * F;
    for (int f = threadIdx.x; f < F; f += blockDim.x) {
        __hip_atomic_fetch_add(&ad[f], hs[f] * w,
                               __ATOMIC_RELAXED, __HIP_MEMORY_SCOPE_AGENT);
    }
}

// ---------------- elementwise kernels ----------------------------------------
__global__ void k_zero(float* p, long n) {
    long i = (long)blockIdx.x * blockDim.x + threadIdx.x;
    if (i < n) p[i] = 0.0f;
}

__global__ void k_bias_relu_dropout(float* __restrict__ h,
                                    const float* __restrict__ bias,
                                    long total, int F)
{
    long i = (long)blockIdx.x * blockDim.x + threadIdx.x;
    if (i >= total) return;
    int f = (int)(i & (F - 1));             // F is power of two (1024)
    float v = h[i] + bias[f];
    v = v > 0.0f ? v : 0.0f;
    // deterministic hash-based Bernoulli(0.5), survivors scaled by 1/(1-p)=2
    unsigned int s = (unsigned int)i;
    s ^= 2747636419u; s *= 2654435769u;
    s ^= s >> 16;     s *= 2654435769u;
    s ^= s >> 16;
    h[i] = (s & 1u) ? (v * 2.0f) : 0.0f;
}

// one wave (32 lanes) per row; F <= 96
__global__ __launch_bounds__(256) void k_logsoftmax_bias(
    const float* __restrict__ agg, const float* __restrict__ bias,
    float* __restrict__ out, int n, int F)
{
    const int wid  = threadIdx.x >> 5;
    const int lane = threadIdx.x & 31;
    const int row  = blockIdx.x * 8 + wid;
    if (row >= n) return;
    const float* x = agg + (long)row * F;

    float vals[3];
    float m = -__builtin_inff();
    #pragma unroll
    for (int j = 0; j < 3; ++j) {
        const int f = lane + j * 32;
        const float v = (f < F) ? (x[f] + bias[f]) : -__builtin_inff();
        vals[j] = v;
        m = fmaxf(m, v);
    }
    #pragma unroll
    for (int o = 16; o > 0; o >>= 1) m = fmaxf(m, __shfl_xor(m, o, 32));

    float s = 0.0f;
    #pragma unroll
    for (int j = 0; j < 3; ++j) {
        const int f = lane + j * 32;
        if (f < F) s += __expf(vals[j] - m);
    }
    #pragma unroll
    for (int o = 16; o > 0; o >>= 1) s += __shfl_xor(s, o, 32);

    const float lse = m + __logf(s);
    #pragma unroll
    for (int j = 0; j < 3; ++j) {
        const int f = lane + j * 32;
        if (f < F) out[(long)row * F + f] = vals[j] - lse;
    }
}

// ---------------- launch -----------------------------------------------------
extern "C" void kernel_launch(void* const* d_in, const int* in_sizes, int n_in,
                              void* d_out, int out_size, void* d_ws, size_t ws_size,
                              hipStream_t stream)
{
    const float*     x     = (const float*)d_in[0];
    const long long* edges = (const long long*)d_in[1];   // int64 [2, E] flat
    const float*     W1    = (const float*)d_in[2];
    const float*     b1    = (const float*)d_in[3];
    const float*     W2    = (const float*)d_in[4];
    const float*     b2    = (const float*)d_in[5];
    float*           out   = (float*)d_out;

    // workspace carve-up (256B aligned)
    char* ws = (char*)d_ws;
    size_t off = 0;
    auto carve = [&](size_t bytes) -> void* {
        void* p = ws + off;
        off = (off + bytes + 255) & ~(size_t)255;
        return p;
    };
    float* dinv = (float*)carve((size_t)NN * 4);             // deg -> dinv in place
    float* norm = (float*)carve((size_t)(EE + NN) * 4);
    float* h1   = (float*)carve((size_t)NN * FH * 4);        // x@W1 ; reused as h2
    float* agg1 = (float*)carve((size_t)NN * FH * 4);
    float* agg2 = (float*)carve((size_t)NN * FOUT * 4);
    float* h2   = h1;                                        // alias: h1 dead by then

    const int T = 256;
    // 1) degrees + symmetric norm
    k_deg_init<<<(NN + T - 1) / T, T, 0, stream>>>(dinv, NN);
    k_deg_acc <<<(EE + T - 1) / T, T, 0, stream>>>(edges, dinv, EE);
    k_dinv    <<<(NN + T - 1) / T, T, 0, stream>>>(dinv, NN);
    k_norm    <<<(EE + NN + T - 1) / T, T, 0, stream>>>(edges, dinv, norm, EE, NN);

    // 2) h1 = x @ W1   (WMMA bf16, f32 accumulate); grid x = N-tiles for L2 reuse
    {
        dim3 grid((FH + BN - 1) / BN, (NN + BM - 1) / BM);
        gemm_wmma_bf16<<<grid, T, 0, stream>>>(x, W1, h1, NN, FIN, FH);
    }

    // 3) agg1 = scatter_add(norm * h1[src] -> dst)
    {
        long tot = (long)NN * FH;
        k_zero<<<(unsigned)((tot + T - 1) / T), T, 0, stream>>>(agg1, tot);
        k_scatter_vec4<<<EE + NN, T, 0, stream>>>(h1, edges, norm, agg1, EE, NN);
        // 4) bias + relu + dropout (in place)
        k_bias_relu_dropout<<<(unsigned)((tot + T - 1) / T), T, 0, stream>>>(agg1, b1, tot, FH);
    }

    // 5) h2 = agg1 @ W2
    {
        dim3 grid((FOUT + BN - 1) / BN, (NN + BM - 1) / BM);
        gemm_wmma_bf16<<<grid, T, 0, stream>>>(agg1, W2, h2, NN, FH, FOUT);
    }

    // 6) agg2 = scatter_add(norm * h2[src] -> dst)
    {
        long tot = (long)NN * FOUT;
        k_zero<<<(unsigned)((tot + T - 1) / T), T, 0, stream>>>(agg2, tot);
        k_scatter<<<EE + NN, 128, 0, stream>>>(h2, edges, norm, agg2, EE, NN, FOUT);
    }

    // 7) out = log_softmax(agg2 + b2)
    k_logsoftmax_bias<<<(NN + 7) / 8, 256, 0, stream>>>(agg2, b2, out, NN, FOUT);
}
